// MultiDynamicMamba_83897891160360
// MI455X (gfx1250) — matrix operand order repaired and checked by reading, not compiled, not run on hardware
//
#include <hip/hip_runtime.h>
#include <math.h>

// ---------------- problem constants ----------------
#define BB 4
#define HH 56
#define WW 56
#define LL (HH*WW)          // 3136
#define DM 192
#define DIN 384
#define DIN2 768
#define NST 16
#define DCONV 4
#define DTR 12
#define RR 44               // DT_RANK + 2*N_STATE
#define RRP 48              // RR padded to WMMA tiles
#define REDC 48
#define NDIR 4

typedef __attribute__((ext_vector_type(16))) _Float16 v16h;
typedef __attribute__((ext_vector_type(8)))  float    v8f;
typedef __attribute__((ext_vector_type(4)))  unsigned int u32x4;
typedef __attribute__((ext_vector_type(4)))  int      i32x4;
typedef __attribute__((ext_vector_type(8)))  int      i32x8;

// scan-order s -> original spatial index l, for each direction
__device__ __forceinline__ int perm_idx(int dir, int s) {
  if (dir == 0) return s;
  if (dir == 1) return LL - 1 - s;
  int t = (dir == 3) ? (LL - 1 - s) : s;
  int j  = t % 7; t /= 7;
  int i  = t % 7; t /= 7;
  int gw = t % 8; t /= 8;
  int gh = t;
  return (gh * 7 + i) * WW + gw * 7 + j;
}

__device__ __forceinline__ float siluf(float x)     { return x / (1.f + __expf(-x)); }
__device__ __forceinline__ float sigmf(float x)     { return 1.f / (1.f + __expf(-x)); }
__device__ __forceinline__ float softplusf_(float x){ return (x > 20.f) ? x : log1pf(__expf(x)); }

// A-fragment: 16x32 f16 from row-major f32, contiguous-K layout, NO guards.
// lane 0-15: M=lane, K in {0..15}+k0 ; lane 16-31: same M, K in {16..31}+k0
__device__ __forceinline__ v16h frag_a_rowmajor(const float* __restrict__ A, int lda,
                                                int m0, int k0) {
  int lane = threadIdx.x & 31;
  const float* p = A + (size_t)(m0 + (lane & 15)) * lda + (k0 + (lane >> 4) * 16);
  float4 a0 = ((const float4*)p)[0];
  float4 a1 = ((const float4*)p)[1];
  float4 b0 = ((const float4*)(p + 8))[0];
  float4 b1 = ((const float4*)(p + 8))[1];
  v16h a;
  a[0]=(_Float16)a0.x; a[1]=(_Float16)a0.y; a[2]=(_Float16)a0.z; a[3]=(_Float16)a0.w;
  a[4]=(_Float16)a1.x; a[5]=(_Float16)a1.y; a[6]=(_Float16)a1.z; a[7]=(_Float16)a1.w;
  a[8]=(_Float16)b0.x; a[9]=(_Float16)b0.y; a[10]=(_Float16)b0.z; a[11]=(_Float16)b0.w;
  a[12]=(_Float16)b1.x; a[13]=(_Float16)b1.y; a[14]=(_Float16)b1.z; a[15]=(_Float16)b1.w;
  return a;
}

// B-fragment: 32x16 f16 where source is row-major [n][k] (K contiguous per lane).
// element i: K = k0 + (lane>>4)*16 + i, N = n0 + (lane&15)
__device__ __forceinline__ v16h frag_b_krows(const float* __restrict__ Wm, int ldb,
                                             int n0, int k0) {
  int lane = threadIdx.x & 31;
  const float* p = Wm + (size_t)(n0 + (lane & 15)) * ldb + (k0 + (lane >> 4) * 16);
  float4 c0 = ((const float4*)p)[0];
  float4 c1 = ((const float4*)p)[1];
  float4 c2 = ((const float4*)p)[2];
  float4 c3 = ((const float4*)p)[3];
  v16h bf;
  bf[0]=(_Float16)c0.x;  bf[1]=(_Float16)c0.y;  bf[2]=(_Float16)c0.z;  bf[3]=(_Float16)c0.w;
  bf[4]=(_Float16)c1.x;  bf[5]=(_Float16)c1.y;  bf[6]=(_Float16)c1.z;  bf[7]=(_Float16)c1.w;
  bf[8]=(_Float16)c2.x;  bf[9]=(_Float16)c2.y;  bf[10]=(_Float16)c2.z; bf[11]=(_Float16)c2.w;
  bf[12]=(_Float16)c3.x; bf[13]=(_Float16)c3.y; bf[14]=(_Float16)c3.z; bf[15]=(_Float16)c3.w;
  return bf;
}

// ---------------- kernels ----------------

__global__ void mask_kernel(const float* __restrict__ x, const float* __restrict__ mw,
                            const float* __restrict__ mb, float* __restrict__ mask) {
  int idx = blockIdx.x * blockDim.x + threadIdx.x;
  if (idx >= BB * LL) return;
  const float* xr = x + (size_t)idx * DM;
  float acc = mb[0];
  for (int k = 0; k < DM; ++k) acc += xr[k] * mw[k];
  mask[idx] = (acc > 0.f) ? 1.f : 0.f;   // sigmoid(acc) > 0.5  <=>  acc > 0
}

// zero-padded f16-friendly copy of x_proj_w: (NDIR, 48, 384)
__global__ void pad_xpw_kernel(const float* __restrict__ xpw, float* __restrict__ xpw_pad) {
  int idx = blockIdx.x * blockDim.x + threadIdx.x;
  if (idx >= NDIR * RRP * DIN) return;
  int c = idx % DIN;
  int r = (idx / DIN) % RRP;
  int d = idx / (DIN * RRP);
  xpw_pad[idx] = (r < RR) ? xpw[((size_t)d * RR + r) * DIN + c] : 0.f;
}

// xz[m,n] = sum_k x[m,k] * in_proj_w[n,k]   (M=12544, K=192, N=768)
__global__ void __launch_bounds__(128) gemm_inproj(const float* __restrict__ X,
                                                   const float* __restrict__ W,
                                                   float* __restrict__ XZ) {
  int wid = threadIdx.x >> 5, lane = threadIdx.x & 31;
  int m0 = blockIdx.y * 16;
  int n0 = (blockIdx.x * 4 + wid) * 16;
  v8f acc = {};
#pragma unroll
  for (int kt = 0; kt < DM / 32; ++kt) {
    v16h a  = frag_a_rowmajor(X, DM, m0, kt * 32);
    v16h bf = frag_b_krows(W, DM, n0, kt * 32);
    acc = __builtin_amdgcn_wmma_f32_16x16x32_f16(false, a, false, bf,
                                                 (short)0, acc, false, false);
  }
  int n  = n0 + (lane & 15);
  int mb = m0 + ((lane >> 4) * 8);
#pragma unroll
  for (int r = 0; r < 8; ++r) XZ[(size_t)(mb + r) * DIN2 + n] = acc[r];
}

// fused: gather xz in scan order, apply mask, causal depthwise conv4 + bias, silu, mask
__global__ void __launch_bounds__(256) conv_kernel(const float* __restrict__ XZ,
                                                   const float* __restrict__ mask,
                                                   const float* __restrict__ cw,
                                                   const float* __restrict__ cb,
                                                   float* __restrict__ conv, int dir) {
  __shared__ float sh[256 + 3];
  int c = blockIdx.y, b = blockIdx.z;
  int s0 = blockIdx.x * 256, t = threadIdx.x;
  int s = s0 + t;
  float xkv = 0.f;
  if (s < LL) {
    int l = perm_idx(dir, s);
    xkv = XZ[((size_t)b * LL + l) * DIN2 + c] * mask[b * LL + l];
  }
  sh[t + 3] = xkv;
  if (t < 3) {
    int sp = s0 - 3 + t;
    float v = 0.f;
    if (sp >= 0) {
      int l = perm_idx(dir, sp);
      v = XZ[((size_t)b * LL + l) * DIN2 + c] * mask[b * LL + l];
    }
    sh[t] = v;
  }
  __syncthreads();
  if (s >= LL) return;
  float acc = cb[c] + cw[c*4+0]*sh[t] + cw[c*4+1]*sh[t+1]
                    + cw[c*4+2]*sh[t+2] + cw[c*4+3]*sh[t+3];
  int l = perm_idx(dir, s);
  conv[((size_t)b * DIN + c) * LL + s] = siluf(acc) * mask[b * LL + l];
}

// x_dbl[b,r,s] = sum_c xpw_pad[r,c] * conv[b,c,s]  (M=48, K=384, N=3136)
// B-panel (384 rows x 16 cols, row stride LL) staged into LDS by the Tensor Data Mover.
__global__ void __launch_bounds__(96) gemm_xdbl(const float* __restrict__ conv,
                                                const float* __restrict__ xpw_pad,
                                                float* __restrict__ xdbl) {
  extern __shared__ float bpanel[];   // DIN*16 floats = 24 KB, bpanel[k*16 + j]
  int wid = threadIdx.x >> 5, lane = threadIdx.x & 31;
  int b  = blockIdx.z;
  int n0 = blockIdx.x * 16;
  int m0 = wid * 16;                  // waves 0..2 cover rows 0..47

  if (wid == 0) {
    // ---- Tensor DMA descriptor (ISA 8.3 / 8.4), 2D tile load ----
    unsigned long long gaddr =
        (unsigned long long)(const void*)(conv + ((size_t)b * DIN * LL + n0));
    unsigned lds_off = __builtin_amdgcn_groupstaticsize();  // dynamic-LDS base
    u32x4 g0;
    g0[0] = 1u;                                            // count=1, user descriptor
    g0[1] = lds_off;                                       // lds_addr (bytes)
    g0[2] = (unsigned)(gaddr & 0xFFFFFFFFu);               // global_addr[31:0]
    g0[3] = (unsigned)((gaddr >> 32) & 0x1FFFFFFu) | (2u << 30);  // addr[56:32] | type=2
    i32x8 g1;
    g1[0] = (int)(2u << 16);                               // wg_mask=0, data_size=4B
    g1[1] = (int)(((unsigned)LL & 0xFFFFu) << 16);         // tensor_dim0[15:0]
    g1[2] = (int)((((unsigned)LL >> 16) & 0xFFFFu) |
                  (((unsigned)DIN & 0xFFFFu) << 16));      // tensor_dim0 hi | tensor_dim1 lo
    g1[3] = (int)((((unsigned)DIN >> 16) & 0xFFFFu) |
                  (16u << 16));                            // tensor_dim1 hi | tile_dim0=16
    g1[4] = (int)((unsigned)DIN);                          // tile_dim1=384, tile_dim2=0
    g1[5] = (int)((unsigned)LL);                           // tensor_dim0_stride[31:0]
    g1[6] = 0;                                             // stride hi | dim1_stride lo
    g1[7] = 0;
    i32x4 z4 = {0, 0, 0, 0};
#if defined(__clang_major__) && __clang_major__ >= 23
    i32x8 z8 = {0, 0, 0, 0, 0, 0, 0, 0};
    __builtin_amdgcn_tensor_load_to_lds(g0, g1, z4, z4, z8, 0);
#else
    __builtin_amdgcn_tensor_load_to_lds(g0, g1, z4, z4, 0);
#endif
    __builtin_amdgcn_s_wait_tensorcnt(0);
  }
  __syncthreads();

  v8f acc = {};
  for (int kt = 0; kt < DIN / 32; ++kt) {
    int k0 = kt * 32;
    v16h a = frag_a_rowmajor(xpw_pad, DIN, m0, k0);
    v16h bf;
    int kb = k0 + (lane >> 4) * 16;
    int nn = lane & 15;
#pragma unroll
    for (int i = 0; i < 16; ++i) bf[i] = (_Float16)bpanel[(kb + i) * 16 + nn];
    acc = __builtin_amdgcn_wmma_f32_16x16x32_f16(false, a, false, bf,
                                                 (short)0, acc, false, false);
  }
  int n  = n0 + (lane & 15);
  int mb = m0 + (lane >> 4) * 8;
#pragma unroll
  for (int r = 0; r < 8; ++r) {
    int m = mb + r;
    if (m < RR) xdbl[((size_t)b * RR + m) * LL + n] = acc[r];
  }
}

// fused selective scan: delta GEMV+softplus, 16-state recurrence, D-skip, silu(z) gate
__global__ void __launch_bounds__(DIN) scan_kernel(const float* __restrict__ conv,
                                                   const float* __restrict__ xdbl,
                                                   const float* __restrict__ XZ,
                                                   const float* __restrict__ dtw,
                                                   const float* __restrict__ dtb,
                                                   const float* __restrict__ A_log,
                                                   const float* __restrict__ Dsk,
                                                   float* __restrict__ ys, int dir) {
  __shared__ float xd[RR];
  int b = blockIdx.x, c = threadIdx.x;
  float Ac[NST], h[NST], w[DTR];
#pragma unroll
  for (int n = 0; n < NST; ++n) { Ac[n] = -__expf(A_log[c * NST + n]); h[n] = 0.f; }
#pragma unroll
  for (int r = 0; r < DTR; ++r) w[r] = dtw[c * DTR + r];
  float bias = dtb[c], Dc = Dsk[c];
  const float* convb = conv + ((size_t)b * DIN + c) * LL;
  const float* xdb   = xdbl + (size_t)b * RR * LL;
  for (int s = 0; s < LL; ++s) {
    __syncthreads();
    if (c < RR) xd[c] = xdb[(size_t)c * LL + s];
    __syncthreads();
    float dt = bias;
#pragma unroll
    for (int r = 0; r < DTR; ++r) dt += w[r] * xd[r];
    dt = softplusf_(dt);
    float u = convb[s];
    if (s + 1 < LL) __builtin_prefetch(convb + s + 1, 0, 3);  // global_prefetch_b8
    float du = dt * u, y = 0.f;
#pragma unroll
    for (int n = 0; n < NST; ++n) {
      h[n] = __expf(dt * Ac[n]) * h[n] + du * xd[DTR + n];
      y += h[n] * xd[DTR + NST + n];
    }
    y += Dc * u;
    int l = perm_idx(dir, s);
    float z = XZ[((size_t)b * LL + l) * DIN2 + DIN + c];
    ys[((size_t)b * DIN + c) * LL + s] = y * siluf(z);
  }
}

// reverse direction: (b,c,s) -> (b,l,c) with l = perm(s)
__global__ void scatter_kernel(const float* __restrict__ ys, float* __restrict__ yo, int dir) {
  int idx = blockIdx.x * blockDim.x + threadIdx.x;
  if (idx >= BB * DIN * LL) return;
  int s = idx % LL;
  int c = (idx / LL) % DIN;
  int b = idx / (LL * DIN);
  int l = perm_idx(dir, s);
  yo[((size_t)b * LL + l) * DIN + c] = ys[idx];
}

// per-row layernorm; accumulate sum of xn over L via f32 atomics
__global__ void __launch_bounds__(DIN) lnsum_kernel(const float* __restrict__ yo,
                                                    const float* __restrict__ g,
                                                    const float* __restrict__ be,
                                                    float* __restrict__ sum_xn) {
  __shared__ float wsum[12];
  __shared__ float mu_s, rs_s;
  int row = blockIdx.x;           // 0 .. B*L-1
  int b = row / LL, c = threadIdx.x;
  int wid = c >> 5, lane = c & 31;
  float v = yo[(size_t)row * DIN + c];
  float loc = v;
  for (int off = 16; off; off >>= 1) loc += __shfl_down(loc, off, 32);
  if (lane == 0) wsum[wid] = loc;
  __syncthreads();
  if (c == 0) { float s = 0.f; for (int i = 0; i < 12; ++i) s += wsum[i]; mu_s = s / DIN; }
  __syncthreads();
  float d = v - mu_s;
  loc = d * d;
  for (int off = 16; off; off >>= 1) loc += __shfl_down(loc, off, 32);
  if (lane == 0) wsum[wid] = loc;
  __syncthreads();
  if (c == 0) { float s = 0.f; for (int i = 0; i < 12; ++i) s += wsum[i];
                rs_s = rsqrtf(s / DIN + 1e-5f); }
  __syncthreads();
  float xn = d * rs_s * g[c] + be[c];
  atomicAdd(&sum_xn[b * DIN + c], xn);
}

// c_attn = sigmoid(sel(gelu(red(mean_xn))))
__global__ void __launch_bounds__(DIN) attn_kernel(const float* __restrict__ sum_xn,
                                                   const float* __restrict__ rw,
                                                   const float* __restrict__ rb,
                                                   const float* __restrict__ sw,
                                                   const float* __restrict__ sb,
                                                   float* __restrict__ cattn) {
  __shared__ float gsh[DIN];
  __shared__ float ggsh[REDC];
  int b = blockIdx.x, c = threadIdx.x;
  gsh[c] = sum_xn[b * DIN + c] / (float)LL;
  __syncthreads();
  if (c < REDC) {
    float acc = rb[c];
    for (int k = 0; k < DIN; ++k) acc += rw[c * DIN + k] * gsh[k];
    ggsh[c] = 0.5f * acc * (1.f + erff(acc * 0.7071067811865475f));  // exact gelu
  }
  __syncthreads();
  float acc = sb[c];
  for (int k = 0; k < REDC; ++k) acc += sw[c * REDC + k] * ggsh[k];
  cattn[b * DIN + c] = sigmf(acc);
}

__global__ void accum_kernel(const float* __restrict__ yo, const float* __restrict__ cattn,
                             float* __restrict__ out_acc) {
  int idx = blockIdx.x * blockDim.x + threadIdx.x;
  if (idx >= BB * LL * DIN) return;
  int c = idx % DIN;
  int b = idx / (LL * DIN);
  out_acc[idx] += yo[idx] * cattn[b * DIN + c];
}

// out[m,n] = sum_c out_acc[m,c]*out_proj_w[n,c] + b[n]   (M=12544, K=384, N=192)
__global__ void __launch_bounds__(128) gemm_out(const float* __restrict__ OA,
                                                const float* __restrict__ W,
                                                const float* __restrict__ bias,
                                                float* __restrict__ out) {
  int wid = threadIdx.x >> 5, lane = threadIdx.x & 31;
  int m0 = blockIdx.y * 16;
  int n0 = (blockIdx.x * 4 + wid) * 16;
  v8f acc = {};
  for (int kt = 0; kt < DIN / 32; ++kt) {
    v16h a  = frag_a_rowmajor(OA, DIN, m0, kt * 32);
    v16h bf = frag_b_krows(W, DIN, n0, kt * 32);
    acc = __builtin_amdgcn_wmma_f32_16x16x32_f16(false, a, false, bf,
                                                 (short)0, acc, false, false);
  }
  int n  = n0 + (lane & 15);
  int mb = m0 + (lane >> 4) * 8;
  float bv = bias[n];
#pragma unroll
  for (int r = 0; r < 8; ++r) out[(size_t)(mb + r) * DM + n] = acc[r] + bv;
}

// ---------------- launcher ----------------
extern "C" void kernel_launch(void* const* d_in, const int* in_sizes, int n_in,
                              void* d_out, int out_size, void* d_ws, size_t ws_size,
                              hipStream_t stream) {
  (void)in_sizes; (void)n_in; (void)out_size; (void)ws_size;
  const float* x          = (const float*)d_in[0];
  const float* in_proj_w  = (const float*)d_in[1];
  const float* conv_w     = (const float*)d_in[2];
  const float* conv_b     = (const float*)d_in[3];
  const float* x_proj_w   = (const float*)d_in[4];
  const float* dt_w       = (const float*)d_in[5];
  const float* dt_b       = (const float*)d_in[6];
  const float* A_log      = (const float*)d_in[7];
  const float* D_skip     = (const float*)d_in[8];
  const float* out_proj_w = (const float*)d_in[9];
  const float* out_proj_b = (const float*)d_in[10];
  const float* ln_g       = (const float*)d_in[11];
  const float* ln_b       = (const float*)d_in[12];
  const float* red_w      = (const float*)d_in[13];
  const float* red_b      = (const float*)d_in[14];
  const float* sel_w      = (const float*)d_in[15];
  const float* sel_b      = (const float*)d_in[16];
  const float* mask_w     = (const float*)d_in[17];
  const float* mask_b     = (const float*)d_in[18];

  float* ws = (float*)d_ws;
  size_t off = 0;
  float* mask    = ws + off; off += (size_t)BB * LL;
  float* xz      = ws + off; off += (size_t)BB * LL * DIN2;
  float* conv    = ws + off; off += (size_t)BB * DIN * LL;
  float* xdbl    = ws + off; off += (size_t)BB * RR * LL;
  float* yscan   = ws + off; off += (size_t)BB * DIN * LL;
  float* yorig   = ws + off; off += (size_t)BB * LL * DIN;
  float* sumxn   = ws + off; off += (size_t)BB * DIN;
  float* cattn   = ws + off; off += (size_t)BB * DIN;
  float* outacc  = ws + off; off += (size_t)BB * LL * DIN;
  float* xpw_pad = ws + off; off += (size_t)NDIR * RRP * DIN;   // total ~118 MB

  hipMemsetAsync(outacc, 0, (size_t)BB * LL * DIN * sizeof(float), stream);

  mask_kernel<<<dim3((BB * LL + 255) / 256), 256, 0, stream>>>(x, mask_w, mask_b, mask);
  pad_xpw_kernel<<<dim3((NDIR * RRP * DIN + 255) / 256), 256, 0, stream>>>(x_proj_w, xpw_pad);
  gemm_inproj<<<dim3(DIN2 / 64, BB * LL / 16), 128, 0, stream>>>(x, in_proj_w, xz);

  for (int d = 0; d < NDIR; ++d) {
    conv_kernel<<<dim3((LL + 255) / 256, DIN, BB), 256, 0, stream>>>(
        xz, mask, conv_w, conv_b, conv, d);
    gemm_xdbl<<<dim3(LL / 16, 1, BB), 96, DIN * 16 * sizeof(float), stream>>>(
        conv, xpw_pad + (size_t)d * RRP * DIN, xdbl);
    scan_kernel<<<dim3(BB), DIN, 0, stream>>>(
        conv, xdbl, xz, dt_w + (size_t)d * DIN * DTR, dt_b + d * DIN,
        A_log, D_skip, yscan, d);
    scatter_kernel<<<dim3((BB * DIN * LL + 255) / 256), 256, 0, stream>>>(yscan, yorig, d);
    hipMemsetAsync(sumxn, 0, BB * DIN * sizeof(float), stream);
    lnsum_kernel<<<dim3(BB * LL), DIN, 0, stream>>>(yorig, ln_g, ln_b, sumxn);
    attn_kernel<<<dim3(BB), DIN, 0, stream>>>(sumxn, red_w, red_b, sel_w, sel_b, cattn);
    accum_kernel<<<dim3((BB * LL * DIN + 255) / 256), 256, 0, stream>>>(yorig, cattn, outacc);
  }

  gemm_out<<<dim3(DM / 64, BB * LL / 16), 128, 0, stream>>>(
      outacc, out_proj_w, out_proj_b, (float*)d_out);
}
